// SinkhornKnopp_imb_63788854280261
// MI455X (gfx1250) — compile-verified, matrix operand order checked by reading, not run on hardware
//
#include <hip/hip_runtime.h>
#include <hip/hip_bf16.h>

typedef __bf16 bf16_t;
typedef __attribute__((ext_vector_type(16))) __bf16 v16bf;
typedef __attribute__((ext_vector_type(8)))  float  v8f;
typedef __attribute__((ext_vector_type(4)))  float  v4f;
typedef __attribute__((ext_vector_type(4)))  unsigned int v4u;

#define DIM   512
#define KC    1000
#define KPAD  1024
#define BROWS 65536
#define TEMP_INV 10.0f      // 1/TEMPERATURE
#define INV_EPS  20.0f      // 1/EPSILON

union FragAB {
    v16bf bf;
    v4u   q[2];
};

// ---------------------------------------------------------------------------
// Kernel 1: L2-normalize centroids -> bf16, padded to KPAD rows (zeros).
// One wave (32 lanes) per row; 512 elements -> 16 per lane.
// ---------------------------------------------------------------------------
__global__ __launch_bounds__(32)
void centroid_norm_kernel(const float* __restrict__ cen, bf16_t* __restrict__ cN) {
    const int row  = blockIdx.x;      // 0..KPAD-1
    const int lane = threadIdx.x;     // 0..31
    bf16_t out[16];
    if (row < KC) {
        const float* src = cen + (size_t)row * DIM;
        float vals[16];
        float ss = 0.f;
#pragma unroll
        for (int i = 0; i < 16; ++i) {
            float x = src[lane + i * 32];
            vals[i] = x;
            ss += x * x;
        }
#pragma unroll
        for (int off = 16; off > 0; off >>= 1)
            ss += __shfl_xor(ss, off, 32);
        float sc = 1.0f / fmaxf(sqrtf(ss), 1e-12f);
#pragma unroll
        for (int i = 0; i < 16; ++i) out[i] = (bf16_t)(vals[i] * sc);
    } else {
#pragma unroll
        for (int i = 0; i < 16; ++i) out[i] = (bf16_t)0.0f;
    }
    bf16_t* dst = cN + (size_t)row * DIM;
#pragma unroll
    for (int i = 0; i < 16; ++i) dst[lane + i * 32] = out[i];
}

// ---------------------------------------------------------------------------
// Kernel 2: fused  normalize(A-rows) -> WMMA GEMM (bf16) -> softmax ->
//           Q = exp(log(softmax+1e-8)/eps)  written to d_out [B,KC] fp32.
// Block = 4 waves (128 thr), owns 16 embedding rows x all KPAD columns.
// Wave w computes N-tiles [w*16, w*16+16)  -> 16 accumulator tiles.
// ---------------------------------------------------------------------------
__global__ __launch_bounds__(128)
void sim_softmax_q_kernel(const float* __restrict__ emb,
                          const bf16_t* __restrict__ cN,
                          float* __restrict__ Q) {
    // A tile in bf16, row stride padded (+8) so ds_load_b128 rows hit
    // different bank groups; 16-byte aligned.
    __shared__ __align__(16) bf16_t lds_a[16][DIM + 8];
    __shared__ float lds_s[16][KPAD];      // sim*10, fp32

    const int tid  = threadIdx.x;
    const int wave = tid >> 5;             // 0..3
    const int lane = tid & 31;
    const int row0 = blockIdx.x * 16;      // base embedding row

    // ---- Phase 1: load + L2-normalize 16 rows of embeddings into LDS (bf16)
    for (int r = wave * 4; r < wave * 4 + 4; ++r) {
        const float* src = emb + (size_t)(row0 + r) * DIM;
        float vals[16];
        float ss = 0.f;
#pragma unroll
        for (int i = 0; i < 16; ++i) {
            float x = src[lane + i * 32];
            vals[i] = x;
            ss += x * x;
        }
#pragma unroll
        for (int off = 16; off > 0; off >>= 1)
            ss += __shfl_xor(ss, off, 32);
        float sc = 1.0f / fmaxf(sqrtf(ss), 1e-12f);
#pragma unroll
        for (int i = 0; i < 16; ++i)
            lds_a[r][lane + i * 32] = (bf16_t)(vals[i] * sc);
    }
    __syncthreads();

    // ---- Phase 2: WMMA GEMM. ISA 16-bit A/B fragment layout (sec 7.12.2):
    // lane = hi*16 + r ; halfs[0..7] = K(k0 + hi*8 .. +7),
    //                    halfs[8..15] = K(k0+16+hi*8 .. +7)
    const int r  = lane & 15;
    const int hi = lane >> 4;

    v8f acc[16] = {};   // 16 tiles x 8 VGPRs

    for (int ks = 0; ks < DIM / 32; ++ks) {
        const int k0 = ks * 32;
        FragAB a;
        a.q[0] = *(const v4u*)&lds_a[r][k0 + hi * 8];
        a.q[1] = *(const v4u*)&lds_a[r][k0 + 16 + hi * 8];
#pragma unroll
        for (int t = 0; t < 16; ++t) {
            const int n = (wave * 16 + t) * 16 + r;   // cN row = tile column
            const bf16_t* bp = cN + (size_t)n * DIM + k0 + hi * 8;
            FragAB b;
            b.q[0] = *(const v4u*)bp;
            b.q[1] = *(const v4u*)(bp + 16);
            acc[t] = __builtin_amdgcn_wmma_f32_16x16x32_bf16(
                false, a.bf, false, b.bf, (short)0, acc[t], false, false);
        }
    }

    // ---- spill sim*TEMP_INV to LDS.  C/D layout: elem (m = v+8*hi, n = r)
#pragma unroll
    for (int t = 0; t < 16; ++t) {
        const int ncol = wave * 256 + t * 16 + r;
#pragma unroll
        for (int v = 0; v < 8; ++v)
            lds_s[v + hi * 8][ncol] = acc[t][v] * TEMP_INV;
    }
    __syncthreads();

    // ---- Phase 3: masked softmax over k<KC, then Q = exp(log(p+1e-8)/eps).
    // Accumulator VGPRs are dead here; cache the row in 32 registers so the
    // LDS tile is swept once instead of three times.
    for (int rr = wave; rr < 16; rr += 4) {
        float vals[32];
        float m = -3.4e38f;
#pragma unroll
        for (int j = 0; j < 32; ++j) {
            const int k = lane + j * 32;
            float v = (k < KC) ? lds_s[rr][k] : -3.4e38f;
            vals[j] = v;
            m = fmaxf(m, v);
        }
#pragma unroll
        for (int off = 16; off > 0; off >>= 1)
            m = fmaxf(m, __shfl_xor(m, off, 32));

        float s = 0.f;
#pragma unroll
        for (int j = 0; j < 32; ++j) {
            const int k = lane + j * 32;
            float e = (k < KC) ? __expf(vals[j] - m) : 0.f;
            vals[j] = e;
            s += e;
        }
#pragma unroll
        for (int off = 16; off > 0; off >>= 1)
            s += __shfl_xor(s, off, 32);
        const float inv = 1.0f / s;

        float* dst = Q + (size_t)(row0 + rr) * KC;
#pragma unroll
        for (int j = 0; j < 32; ++j) {
            const int k = lane + j * 32;
            if (k < KC) {
                float p = vals[j] * inv;
                dst[k] = __expf(__logf(p + 1e-8f) * INV_EPS);
            }
        }
    }
}

// ---------------------------------------------------------------------------
// Sinkhorn helpers (bandwidth-bound; global scale factors cancel analytically)
// ---------------------------------------------------------------------------
__global__ __launch_bounds__(256)
void zero_kernel(float* __restrict__ p, int n) {
    int i = blockIdx.x * 256 + threadIdx.x;
    if (i < n) p[i] = 0.f;
}

// classsum[k] = sum_b Q[b][k].  Rows are 4000B (16B aligned) -> float4 loads;
// 250 active lanes cover K=1000.  512 blocks x 128 rows for high MLP.
__global__ __launch_bounds__(256)
void class_sum_kernel(const float* __restrict__ Q, float* __restrict__ classsum) {
    const int k4 = threadIdx.x;                 // float4 index
    if (k4 * 4 >= KC) return;
    const int b0 = blockIdx.x * (BROWS / 512);
    v4f part = {0.f, 0.f, 0.f, 0.f};
    for (int b = b0; b < b0 + (BROWS / 512); ++b) {
        const v4f* rowp = (const v4f*)(Q + (size_t)b * KC);
        part += rowp[k4];
    }
    atomicAdd(&classsum[k4 * 4 + 0], part.x);
    atomicAdd(&classsum[k4 * 4 + 1], part.y);
    atomicAdd(&classsum[k4 * 4 + 2], part.z);
    atomicAdd(&classsum[k4 * 4 + 3], part.w);
}

// ratio[k] = cls_dist[k] / classsum[k]  (in place; removes 65M divides from
// the big pass below)
__global__ __launch_bounds__(256)
void ratio_kernel(float* __restrict__ classsum, const float* __restrict__ cls_dist) {
    int k = blockIdx.x * 256 + threadIdx.x;
    if (k < KC) classsum[k] = cls_dist[k] / classsum[k];
}

// Q[b][k] = (Q[b][k]*ratio[k]) / sum_k(Q[b][k]*ratio[k])  ; 1 wave per row.
__global__ __launch_bounds__(128)
void row_update_kernel(float* __restrict__ Q,
                       const float* __restrict__ ratio) {
    const int wave = threadIdx.x >> 5;
    const int lane = threadIdx.x & 31;
    const size_t b = (size_t)blockIdx.x * 4 + wave;
    float* rowp = Q + b * KC;

    float vals[32];
    float s = 0.f;
#pragma unroll
    for (int j = 0; j < 32; ++j) {
        const int k = lane + j * 32;
        float v = 0.f;
        if (k < KC) v = rowp[k] * ratio[k];
        vals[j] = v;
        s += v;
    }
#pragma unroll
    for (int off = 16; off > 0; off >>= 1)
        s += __shfl_xor(s, off, 32);
    const float inv = 1.0f / s;
#pragma unroll
    for (int j = 0; j < 32; ++j) {
        const int k = lane + j * 32;
        if (k < KC) rowp[k] = vals[j] * inv;
    }
}

// ---------------------------------------------------------------------------
extern "C" void kernel_launch(void* const* d_in, const int* in_sizes, int n_in,
                              void* d_out, int out_size, void* d_ws, size_t ws_size,
                              hipStream_t stream) {
    const float* emb = (const float*)d_in[0];   // [65536,512]
    const float* cen = (const float*)d_in[1];   // [1000,512]
    const float* cls = (const float*)d_in[2];   // [1000,1]
    float* Q = (float*)d_out;                   // [65536,1000]

    bf16_t* cN      = (bf16_t*)d_ws;                                  // 1 MB
    float*  classum = (float*)((char*)d_ws + (size_t)KPAD * DIM * 2); // 4 KB

    centroid_norm_kernel<<<KPAD, 32, 0, stream>>>(cen, cN);
    sim_softmax_q_kernel<<<BROWS / 16, 128, 0, stream>>>(emb, cN, Q);

    for (int it = 0; it < 3; ++it) {
        zero_kernel<<<(KC + 255) / 256, 256, 0, stream>>>(classum, KC);
        class_sum_kernel<<<512, 256, 0, stream>>>(Q, classum);
        ratio_kernel<<<(KC + 255) / 256, 256, 0, stream>>>(classum, cls);
        row_update_kernel<<<BROWS / 4, 128, 0, stream>>>(Q, classum);
    }
}
